// ContrastiveLossInBatch_26843545600181
// MI455X (gfx1250) — matrix operand order repaired and checked by reading, not compiled, not run on hardware
//
#include <hip/hip_runtime.h>
#include <hip/hip_bf16.h>
#include <cstdint>

// Contrastive (InfoNCE-style) loss, fused GEMM + row-wise logsumexp.
//   sim = (q @ k^T) / 0.07 ; loss = mean_i( logsumexp_j sim[i,j] - sim[i,i] )
// N=8192, C=256, fp32 in, scalar fp32 out.
//
// bf16x3 split-precision WMMA GEMM (qh*kh + qh*kl + ql*kh), online logsumexp
// in registers, k tiles staged via gfx1250 GLOBAL_LOAD_ASYNC_TO_LDS_B128 with
// double buffering (ASYNCcnt-paced pipeline). Workspace: 4 bf16 planes = 16 MB.

#define N_ROWS 8192
#define C_DIM  256
#define INV_T  (1.0f / 0.07f)

#define WG_WAVES 4
#define NT (N_ROWS / 16)            // 512 column tiles
#define COLSTRIDE 264               // halfs per column in LDS (512B + 16B pad)
#define TERM_STRIDE (16 * COLSTRIDE)
#define BUF_HALFS (2 * TERM_STRIDE) // hi + lo planes, one tile
#define LDS_HALFS (2 * BUF_HALFS)   // double buffered

typedef __attribute__((ext_vector_type(16))) __bf16 v16bf;
typedef __attribute__((ext_vector_type(8)))  float  v8f;

union Frag16 { uint4 u[2]; v16bf v; };

#if __has_builtin(__builtin_amdgcn_s_wait_asynccnt)
#define WAIT_ASYNC(n) __builtin_amdgcn_s_wait_asynccnt(n)
#else
#define WAIT_ASYNC(n) asm volatile("s_wait_asynccnt %0" ::"i"(n) : "memory")
#endif

__device__ __forceinline__ uint32_t lds_offset_of(void* p) {
    // generic -> LDS address space, then to the 32-bit LDS byte offset the
    // async-to-LDS instructions take in their VDST operand.
    return (uint32_t)(__UINTPTR_TYPE__)(__attribute__((address_space(3))) void*)p;
}

__device__ __forceinline__ void async_copy_b128(void* lds_dst, const void* gsrc) {
    asm volatile("global_load_async_to_lds_b128 %0, %1, off"
                 :: "v"(lds_offset_of(lds_dst)), "v"(gsrc)
                 : "memory");
}

__device__ __forceinline__ unsigned short f32_to_bf16_rn(float x) {
    union { float f; uint32_t u; } c; c.f = x;
    uint32_t r = c.u + 0x7FFFu + ((c.u >> 16) & 1u);   // round-to-nearest-even
    return (unsigned short)(r >> 16);
}
__device__ __forceinline__ float bf16_bits_to_f32(unsigned short h) {
    union { uint32_t u; float f; } c; c.u = ((uint32_t)h) << 16;
    return c.f;
}

// ---- Pass 0: split fp32 q,k into bf16 hi/lo planes; zero the output ----
__global__ void split_bf16_kernel(const float* __restrict__ q,
                                  const float* __restrict__ k,
                                  unsigned short* __restrict__ ws,
                                  float* __restrict__ out) {
    const int NC = N_ROWS * C_DIM;
    int idx = blockIdx.x * blockDim.x + threadIdx.x;
    if (idx == 0) out[0] = 0.0f;                 // stream-ordered before pass 1
    if (idx >= 2 * NC) return;
    const float* src = (idx < NC) ? q : k;
    int base = (idx < NC) ? 0 : 2 * NC;          // q planes then k planes
    int e = (idx < NC) ? idx : idx - NC;
    float x = src[e];
    unsigned short hi = f32_to_bf16_rn(x);
    unsigned short lo = f32_to_bf16_rn(x - bf16_bits_to_f32(hi));
    ws[base + e]      = hi;
    ws[base + NC + e] = lo;
}

// ---- Pass 1: fused WMMA GEMM + online logsumexp ----
__launch_bounds__(WG_WAVES * 32, 1)
__global__ void nce_loss_kernel(const unsigned short* __restrict__ ws,
                                float* __restrict__ out) {
    const int NC = N_ROWS * C_DIM;
    const unsigned short* qh = ws;
    const unsigned short* ql = ws + NC;
    const unsigned short* kh = ws + 2 * NC;
    const unsigned short* kl = ws + 3 * NC;

    __shared__ __align__(16) unsigned short lds[LDS_HALFS];

    const int tid  = threadIdx.x;
    const int lane = tid & 31;
    const int row_base = blockIdx.x * (WG_WAVES * 16) + (tid >> 5) * 16;

    // ---- Preload A fragments (this wave's 16 q rows, hi+lo, full K=256) ----
    // 16-bit A layout: lanes 0-15 (row=lane): K[k0..k0+7] then K[k0+16..k0+23];
    // lanes 16-31 (row=lane-16): K[k0+8..k0+15] then K[k0+24..k0+31].
    Frag16 a_hi[8], a_lo[8];
    {
        const int r    = row_base + (lane & 15);
        const int koff = (lane & 16) ? 8 : 0;
        const uint4* qh4 = (const uint4*)(qh + r * C_DIM);
        const uint4* ql4 = (const uint4*)(ql + r * C_DIM);
        #pragma unroll
        for (int kk = 0; kk < 8; ++kk) {
            int e0 = (32 * kk + koff) >> 3;      // in uint4 (8-half) units
            a_hi[kk].u[0] = qh4[e0];
            a_hi[kk].u[1] = qh4[e0 + 2];
            a_lo[kk].u[0] = ql4[e0];
            a_lo[kk].u[1] = ql4[e0 + 2];
        }
    }

    // Online logsumexp state per accumulator VGPR slot.
    float mrun[8], srun[8], pos[8];
    #pragma unroll
    for (int v = 0; v < 8; ++v) { mrun[v] = -3.0e38f; srun[v] = 0.0f; pos[v] = 0.0f; }
    const int diag_tile = row_base >> 4;

    // Async staging: 1024 16B chunks per tile (2 planes * 16 cols * 512B),
    // 8 async b128 copies per thread -> ASYNCcnt += 8 per wave per tile.
    auto async_load_tile = [&](int t, int buf) {
        const int col_base = t * 16;
        unsigned short* base = lds + buf * BUF_HALFS;
        #pragma unroll
        for (int i = 0; i < 8; ++i) {
            int c = tid + 128 * i;
            const unsigned short* src = (c >> 9) ? kl : kh;
            int term = c >> 9;
            int col  = (c >> 5) & 15;
            int off  = (c & 31) * 8;             // halfs
            async_copy_b128(base + term * TERM_STRIDE + col * COLSTRIDE + off,
                            src + (col_base + col) * C_DIM + off);
        }
    };

    async_load_tile(0, 0);
    for (int t = 0; t < NT; ++t) {
        const int cur = t & 1;
        if (t + 1 < NT) {
            async_load_tile(t + 1, cur ^ 1);     // prefetch overlaps compute
            WAIT_ASYNC(8);                        // tile t landed in LDS
        } else {
            WAIT_ASYNC(0);
        }
        __syncthreads();                          // all waves' chunks visible

        // ---- 16x16 tile: two independent WMMA chains over K=256 ----
        v8f acc0 = {}, acc1 = {};
        const unsigned short* bh_base =
            lds + cur * BUF_HALFS + (lane & 15) * COLSTRIDE + ((lane & 16) >> 1);
        const unsigned short* bl_base = bh_base + TERM_STRIDE;
        #pragma unroll
        for (int kk = 0; kk < 8; ++kk) {
            Frag16 bh, bl;
            const uint4* p0 = (const uint4*)(bh_base + 32 * kk);
            bh.u[0] = p0[0]; bh.u[1] = p0[2];
            const uint4* p1 = (const uint4*)(bl_base + 32 * kk);
            bl.u[0] = p1[0]; bl.u[1] = p1[2];
            acc0 = __builtin_amdgcn_wmma_f32_16x16x32_bf16(
                false, a_hi[kk].v, false, bh.v, (short)0, acc0, false, false);
            acc1 = __builtin_amdgcn_wmma_f32_16x16x32_bf16(
                false, a_hi[kk].v, false, bl.v, (short)0, acc1, false, false);
            acc1 = __builtin_amdgcn_wmma_f32_16x16x32_bf16(
                false, a_lo[kk].v, false, bh.v, (short)0, acc1, false, false);
        }

        // C/D layout: slot v = sim(row v, col lane) lanes 0-15;
        //             sim(row v+8, col lane-16) lanes 16-31.
        #pragma unroll
        for (int v = 0; v < 8; ++v) {
            float x  = (acc0[v] + acc1[v]) * INV_T;
            float mn = fmaxf(mrun[v], x);
            srun[v]  = srun[v] * __expf(mrun[v] - mn) + __expf(x - mn);
            mrun[v]  = mn;
        }
        if (t == diag_tile) {
            #pragma unroll
            for (int v = 0; v < 8; ++v) {
                bool is_diag = (lane < 16) ? (lane == v) : (lane == v + 24);
                if (is_diag) pos[v] = (acc0[v] + acc1[v]) * INV_T;
            }
        }
        __syncthreads();          // done reading `cur` before it is re-filled
    }

    // ---- Reduce across the 16 lanes of each half-wave group ----
    float contrib = 0.0f;
    #pragma unroll
    for (int v = 0; v < 8; ++v) {
        float M = mrun[v];
        M = fmaxf(M, __shfl_xor(M, 1));
        M = fmaxf(M, __shfl_xor(M, 2));
        M = fmaxf(M, __shfl_xor(M, 4));
        M = fmaxf(M, __shfl_xor(M, 8));
        float s = srun[v] * __expf(mrun[v] - M);
        s += __shfl_xor(s, 1);
        s += __shfl_xor(s, 2);
        s += __shfl_xor(s, 4);
        s += __shfl_xor(s, 8);
        float P = pos[v];
        P += __shfl_xor(P, 1);
        P += __shfl_xor(P, 2);
        P += __shfl_xor(P, 4);
        P += __shfl_xor(P, 8);
        contrib += (M + __logf(s)) - P;   // uniform within each 16-lane group
    }
    contrib += __shfl_xor(contrib, 16);    // combine the two row groups
    if (lane == 0) atomicAdd(out, contrib * (1.0f / N_ROWS));
}

extern "C" void kernel_launch(void* const* d_in, const int* in_sizes, int n_in,
                              void* d_out, int out_size, void* d_ws, size_t ws_size,
                              hipStream_t stream) {
    const float* q = (const float*)d_in[0];
    const float* k = (const float*)d_in[1];
    float* out = (float*)d_out;                       // single fp32 scalar
    unsigned short* ws = (unsigned short*)d_ws;       // needs 4*N*C*2 = 16 MB

    const int NC = N_ROWS * C_DIM;
    split_bf16_kernel<<<(2 * NC + 255) / 256, 256, 0, stream>>>(q, k, ws, out);
    nce_loss_kernel<<<N_ROWS / (WG_WAVES * 16), WG_WAVES * 32, 0, stream>>>(ws, out);
}
